// MultiHeadSelfAttention_67491116089470
// MI455X (gfx1250) — matrix operand order repaired and checked by reading, not compiled, not run on hardware
//
#include <hip/hip_runtime.h>
#include <hip/hip_bf16.h>

// MHA forward for MI455X (gfx1250, wave32).  All GEMMs via v_wmma_f32_16x16x32_bf16
// (bf16 inputs, fp32 accumulate).  ~69 GFLOP vs ~80MB traffic -> matrix-core bound.

#define D_MODEL   1024
#define NUM_HEADS 16
#define DEPTH     64
#define BATCH     2
#define SEQ       2048

typedef __bf16 bf16;
typedef __attribute__((ext_vector_type(8)))  __bf16 v8bf;
typedef __attribute__((ext_vector_type(16))) __bf16 v16bf;
typedef __attribute__((ext_vector_type(8)))  float  v8f;
typedef __attribute__((ext_vector_type(4)))  int    v4i;

// gfx1250 async global->LDS copy (ASYNCcnt-tracked), guarded so compile can't regress.
#if __has_builtin(__builtin_amdgcn_global_load_async_to_lds_b128) && \
    __has_builtin(__builtin_amdgcn_s_wait_asynccnt)
#define USE_ASYNC_LDS 1
typedef __attribute__((address_space(1))) v4i* gptr_b128;   // global, 16B granule
typedef __attribute__((address_space(3))) v4i* lptr_b128;   // LDS, 16B granule
#else
#define USE_ASYNC_LDS 0
#endif

// ---------------------------------------------------------------------------
// Fragment loaders (wave32): contiguous k order, shared by A and B (not the ISA's table order).
// A-matrix 16x32 bf16: lane m = lane&15, g = lane>>4; element e -> k = e + g*16.
__device__ __forceinline__ v16bf load_a_frag(const bf16* base, int ld) {
  const int lane = threadIdx.x & 31;
  const int m = lane & 15, g = lane >> 4;
  const bf16* p = base + m * ld + g * 16;
  v8bf lo = *(const v8bf*)(p);
  v8bf hi = *(const v8bf*)(p + 8);
  v16bf a;
#pragma unroll
  for (int i = 0; i < 8; ++i) { a[i] = lo[i]; a[i + 8] = hi[i]; }
  return a;
}

// B-matrix 32x16: lane n = lane&15; element e -> k = e + (lane>>4)*16.
// Source must be k-contiguous: element (k,n) at base[n*ld + k] -> one 32B load.
__device__ __forceinline__ v16bf load_b_frag(const bf16* base, int ld) {
  const int lane = threadIdx.x & 31;
  const int n = lane & 15, g = lane >> 4;
  return *(const v16bf*)(base + n * ld + g * 16);
}

__device__ __forceinline__ v8f wmma_bf16(v16bf a, v16bf b, v8f c) {
  // (neg_a, A, neg_b, B, c_mod, C, reuse_a, reuse_b)
  return __builtin_amdgcn_wmma_f32_16x16x32_bf16(false, a, false, b, (short)0, c,
                                                 false, false);
}

// ---------------------------------------------------------------------------
// Kernel 1: QKV projection  X[4096,1024] @ Wqkv[1024,3072] + b, fused RoPE,
// scatter to q/k/v in [b,h,t,d] bf16.  Block tile 128x64, 8 waves, K-step 32.
__global__ void __launch_bounds__(256)
qkv_kernel(const float* __restrict__ X, const float* __restrict__ W,
           const float* __restrict__ Bias,
           bf16* __restrict__ Qo, bf16* __restrict__ Ko, bf16* __restrict__ Vo) {
  __shared__ alignas(128) bf16 ldsA[128 * 32];   // row-major bf16 A tile
  __shared__ alignas(128) bf16 ldsBt[64 * 32];   // k-contiguous B tile

  const int tid = threadIdx.x, wave = tid >> 5, lane = tid & 31;
  const int g = lane >> 4, n = lane & 15;
  const int row0 = blockIdx.x * 128;
  const int col0 = blockIdx.y * 64;

  v8f acc[4];
#pragma unroll
  for (int j = 0; j < 4; ++j) acc[j] = (v8f){0.f,0.f,0.f,0.f,0.f,0.f,0.f,0.f};

  const int ar = tid >> 1;              // A stage: 2 threads per row
  const int ak = (tid & 1) * 16;
  const int bn = tid & 63;              // B stage: column per thread
  const int bk = (tid >> 6) * 8;

  for (int ks = 0; ks < D_MODEL; ks += 32) {
    {  // stage A (fp32 -> bf16), coalesced 64B per thread
      const float* src = X + (size_t)(row0 + ar) * D_MODEL + ks + ak;
      v8bf o0, o1;
#pragma unroll
      for (int i = 0; i < 8; ++i) o0[i] = (bf16)src[i];
#pragma unroll
      for (int i = 0; i < 8; ++i) o1[i] = (bf16)src[8 + i];
      *(v8bf*)&ldsA[ar * 32 + ak]     = o0;
      *(v8bf*)&ldsA[ar * 32 + ak + 8] = o1;
    }
    {  // stage B transposed to k-contiguous (coalesced across lanes per k-row)
      v8bf ob;
#pragma unroll
      for (int i = 0; i < 8; ++i)
        ob[i] = (bf16)W[(size_t)(ks + bk + i) * (3 * D_MODEL) + col0 + bn];
      *(v8bf*)&ldsBt[bn * 32 + bk] = ob;
    }
    __syncthreads();
    // batch all fragment loads, then issue WMMAs back-to-back
    v16bf af = load_a_frag(ldsA + wave * 16 * 32, 32);
    v16bf bfr[4];
#pragma unroll
    for (int j = 0; j < 4; ++j) bfr[j] = load_b_frag(ldsBt + j * 16 * 32, 32);
#pragma unroll
    for (int j = 0; j < 4; ++j) acc[j] = wmma_bf16(af, bfr[j], acc[j]);
    __syncthreads();
  }

  // Epilogue: bias + RoPE (q,k only) + scatter to [b,h,t,d] bf16.
  const int which = col0 >> 10;                 // 0=q 1=k 2=v (uniform per block)
  const int h     = (col0 & (D_MODEL - 1)) >> 6;
  bf16* dst = (which == 0) ? Qo : (which == 1) ? Ko : Vo;
#pragma unroll
  for (int j = 0; j < 4; ++j) {
    const int c = col0 + j * 16 + n;
    const int d = c & (DEPTH - 1);
    const float bias = Bias[c];
    // inv_freq = 10000^(-(2i)/64); 2i = d&~1
    const float inv_freq = __expf(-(float)(d & ~1) * (9.210340372f / 64.f));
#pragma unroll
    for (int r = 0; r < 8; ++r) {
      const int row = row0 + wave * 16 + r + 8 * g;   // C layout: M = r + 8g
      const int t  = row & (SEQ - 1);
      const int bb = row >> 11;
      float val = acc[j][r] + bias;
      if (which < 2) {
        float sn, cs;
        __sincosf((float)t * inv_freq, &sn, &cs);
        float partner = __shfl_xor(val, 1, 32);   // even<->odd dim pair = adjacent lanes
        val = (d & 1) ? (partner * sn + val * cs) : (val * cs - partner * sn);
      }
      dst[(((size_t)bb * NUM_HEADS + h) * SEQ + t) * DEPTH + d] = (bf16)val;
    }
  }
}

// ---------------------------------------------------------------------------
// Kernel 2: flash attention.  One 16-query tile per wave, 8 waves / block share
// LDS-staged 32-key K/V chunks.  Online softmax with lane-local row state.
// K tile is staged with GLOBAL_LOAD_ASYNC_TO_LDS_B128 when the toolchain has it.
__global__ void __launch_bounds__(256)
attn_kernel(const bf16* __restrict__ Q, const bf16* __restrict__ K,
            const bf16* __restrict__ V, bf16* __restrict__ O) {
  __shared__ alignas(128) bf16 ldsK [32 * 64];      // [s][d] row-major (k-contig for S=Q*K^T)
  __shared__ alignas(128) bf16 ldsVt[64 * 32];      // [d][s] transposed (k-contig for P*V)
  __shared__ alignas(128) bf16 ldsP [8][16 * 32];   // per-wave P transpose scratch

  const int tid = threadIdx.x, wave = tid >> 5, lane = tid & 31;
  const int g = lane >> 4, n = lane & 15;
  const int bh = blockIdx.x >> 4, qblk = blockIdx.x & 15;
  const int b = bh >> 4, h = bh & 15;

  const bf16* Qh = Q + (size_t)bh * SEQ * DEPTH;
  const bf16* Kh = K + (size_t)bh * SEQ * DEPTH;
  const bf16* Vh = V + (size_t)bh * SEQ * DEPTH;
  const int t0 = qblk * 128 + wave * 16;

  const v16bf qf0 = load_a_frag(Qh + (size_t)t0 * DEPTH, DEPTH);       // d 0..31
  const v16bf qf1 = load_a_frag(Qh + (size_t)t0 * DEPTH + 32, DEPTH);  // d 32..63

  v8f out[4];
  float m_run[8], l_run[8];
#pragma unroll
  for (int j = 0; j < 4; ++j) out[j] = (v8f){0.f,0.f,0.f,0.f,0.f,0.f,0.f,0.f};
#pragma unroll
  for (int r = 0; r < 8; ++r) { m_run[r] = -1e30f; l_run[r] = 0.f; }

  const int ss = tid >> 3;            // staging: key row
  const int sd = (tid & 7) * 8;       // staging: d offset (8 bf16 = 16B)

  for (int kc = 0; kc < SEQ / 32; ++kc) {
    const size_t goff = (size_t)(kc * 32 + ss) * DEPTH + sd;
    {  // stage K row-major (async when available), V transposed through VGPRs
#if USE_ASYNC_LDS
      __builtin_amdgcn_global_load_async_to_lds_b128(
          (gptr_b128)(void*)(Kh + goff),
          (lptr_b128)(void*)&ldsK[ss * 64 + sd],
          0, 0);
#else
      v8bf kv = *(const v8bf*)(Kh + goff);
      *(v8bf*)&ldsK[ss * 64 + sd] = kv;
#endif
      v8bf vv = *(const v8bf*)(Vh + goff);
#pragma unroll
      for (int i = 0; i < 8; ++i) ldsVt[(sd + i) * 32 + ss] = vv[i];
    }
    if (kc + 1 < SEQ / 32) {  // prefetch next chunk (global_prefetch_b8)
      __builtin_prefetch(Kh + goff + 32 * DEPTH, 0, 3);
      __builtin_prefetch(Vh + goff + 32 * DEPTH, 0, 3);
    }
#if USE_ASYNC_LDS
    __builtin_amdgcn_s_wait_asynccnt(0);
#endif
    __syncthreads();

    // S = Q K^T over 32 keys: batch B-fragment loads, then 4 WMMAs back-to-back
    v16bf kf[4];
    kf[0] = load_b_frag(ldsK + 0,            64);   // keys 0..15, d 0..31
    kf[1] = load_b_frag(ldsK + 32,           64);   // keys 0..15, d 32..63
    kf[2] = load_b_frag(ldsK + 16 * 64,      64);   // keys 16..31, d 0..31
    kf[3] = load_b_frag(ldsK + 16 * 64 + 32, 64);   // keys 16..31, d 32..63
    v8f s0 = (v8f){0.f,0.f,0.f,0.f,0.f,0.f,0.f,0.f};
    v8f s1 = (v8f){0.f,0.f,0.f,0.f,0.f,0.f,0.f,0.f};
    s0 = wmma_bf16(qf0, kf[0], s0);
    s1 = wmma_bf16(qf0, kf[2], s1);
    s0 = wmma_bf16(qf1, kf[1], s0);
    s1 = wmma_bf16(qf1, kf[3], s1);

    // online softmax; row m = r + 8g lives in vgpr r across 16 lanes of this half
    bf16* pw = ldsP[wave];
#pragma unroll
    for (int r = 0; r < 8; ++r) {
      float a0 = s0[r] * 0.125f;                 // 1/sqrt(64)
      float a1 = s1[r] * 0.125f;
      float cm = fmaxf(a0, a1);
      cm = fmaxf(cm, __shfl_xor(cm, 1, 32));
      cm = fmaxf(cm, __shfl_xor(cm, 2, 32));
      cm = fmaxf(cm, __shfl_xor(cm, 4, 32));
      cm = fmaxf(cm, __shfl_xor(cm, 8, 32));
      float mn    = fmaxf(m_run[r], cm);
      float alpha = __expf(m_run[r] - mn);
      float p0 = __expf(a0 - mn), p1 = __expf(a1 - mn);
      float rs = p0 + p1;
      rs += __shfl_xor(rs, 1, 32);
      rs += __shfl_xor(rs, 2, 32);
      rs += __shfl_xor(rs, 4, 32);
      rs += __shfl_xor(rs, 8, 32);
      l_run[r] = l_run[r] * alpha + rs;
      m_run[r] = mn;
#pragma unroll
      for (int j = 0; j < 4; ++j) out[j][r] *= alpha;
      const int m = r + 8 * g;
      pw[m * 32 + n]      = (bf16)p0;            // C-layout -> row-major P tile
      pw[m * 32 + 16 + n] = (bf16)p1;
    }

    // O += P V  (LDS is in-order within a wave: store->load on pw is safe)
    v16bf pf = load_a_frag(pw, 32);
    v16bf vf[4];
#pragma unroll
    for (int j = 0; j < 4; ++j) vf[j] = load_b_frag(ldsVt + j * 16 * 32, 32);
#pragma unroll
    for (int j = 0; j < 4; ++j) out[j] = wmma_bf16(pf, vf[j], out[j]);
    __syncthreads();
  }

  // finalize: divide by l, write [b,t,h*64+d] bf16 (GEMM-ready row-major)
#pragma unroll
  for (int r = 0; r < 8; ++r) {
    const int t = t0 + r + 8 * g;
    const float inv = 1.0f / l_run[r];
    const size_t rowoff = ((size_t)b * SEQ + t) * D_MODEL + h * DEPTH;
#pragma unroll
    for (int j = 0; j < 4; ++j)
      O[rowoff + j * 16 + n] = (bf16)(out[j][r] * inv);
  }
}

// ---------------------------------------------------------------------------
// Kernel 3: output projection  O[4096,1024](bf16) @ Wfc[1024,1024] + b -> fp32.
// A-fragments load straight from global (already bf16 row-major); B staged in LDS.
__global__ void __launch_bounds__(256)
proj_kernel(const bf16* __restrict__ A, const float* __restrict__ W,
            const float* __restrict__ Bias, float* __restrict__ Out) {
  __shared__ alignas(128) bf16 ldsBt[64 * 32];
  const int tid = threadIdx.x, wave = tid >> 5, lane = tid & 31;
  const int g = lane >> 4, n = lane & 15;
  const int row0 = blockIdx.x * 128 + wave * 16;
  const int col0 = blockIdx.y * 64;

  v8f acc[4];
#pragma unroll
  for (int j = 0; j < 4; ++j) acc[j] = (v8f){0.f,0.f,0.f,0.f,0.f,0.f,0.f,0.f};

  const int bn = tid & 63, bk = (tid >> 6) * 8;
  for (int ks = 0; ks < D_MODEL; ks += 32) {
    v8bf ob;
#pragma unroll
    for (int i = 0; i < 8; ++i)
      ob[i] = (bf16)W[(size_t)(ks + bk + i) * D_MODEL + col0 + bn];
    *(v8bf*)&ldsBt[bn * 32 + bk] = ob;
    __syncthreads();
    v16bf af = load_a_frag(A + (size_t)row0 * D_MODEL + ks, D_MODEL);
    v16bf bfr[4];
#pragma unroll
    for (int j = 0; j < 4; ++j) bfr[j] = load_b_frag(ldsBt + j * 16 * 32, 32);
#pragma unroll
    for (int j = 0; j < 4; ++j) acc[j] = wmma_bf16(af, bfr[j], acc[j]);
    __syncthreads();
  }
#pragma unroll
  for (int j = 0; j < 4; ++j) {
    const int c = col0 + j * 16 + n;
    const float bias = Bias[c];
#pragma unroll
    for (int r = 0; r < 8; ++r)
      Out[(size_t)(row0 + r + 8 * g) * D_MODEL + c] = acc[j][r] + bias;
  }
}

// ---------------------------------------------------------------------------
extern "C" void kernel_launch(void* const* d_in, const int* in_sizes, int n_in,
                              void* d_out, int out_size, void* d_ws, size_t ws_size,
                              hipStream_t stream) {
  const float* x     = (const float*)d_in[0];
  const float* w_qkv = (const float*)d_in[1];
  const float* b_qkv = (const float*)d_in[2];
  const float* w_fc  = (const float*)d_in[3];
  const float* b_fc  = (const float*)d_in[4];
  float* out = (float*)d_out;

  const size_t per = (size_t)BATCH * NUM_HEADS * SEQ * DEPTH;  // 4,194,304 elems
  bf16* qb = (bf16*)d_ws;       // 8 MB each
  bf16* kb = qb + per;
  bf16* vb = kb + per;
  bf16* ob = vb + per;          // [b,t,1024] bf16, 8 MB  (total ws use: 32 MB)

  qkv_kernel <<<dim3(32, 48), 256, 0, stream>>>(x, w_qkv, b_qkv, qb, kb, vb);
  attn_kernel<<<dim3(512),    256, 0, stream>>>(qb, kb, vb, ob);
  proj_kernel<<<dim3(32, 16), 256, 0, stream>>>(ob, w_fc, b_fc, out);
}
